// MemoryEfficientAttention_10342281249038
// MI455X (gfx1250) — matrix operand order repaired and checked by reading, not compiled, not run on hardware
//
#include <hip/hip_runtime.h>

typedef _Float16 half_t;
typedef __attribute__((ext_vector_type(16))) _Float16 v16h;
typedef __attribute__((ext_vector_type(8)))  _Float16 v8h;
typedef __attribute__((ext_vector_type(8)))  float    v8f;
typedef __attribute__((ext_vector_type(4)))  float    v4f;

#define DEVINL __device__ __forceinline__

static constexpr int kB = 2;
static constexpr int kT = 2048;
static constexpr int kC = 1024;
static constexpr int kH = 16;
static constexpr int kD = 64;
static constexpr int kChunk = 64;
static constexpr float kNegInf = -1.0e30f;

// ---------------- WMMA / async helpers ----------------

DEVINL v8f wmma_f16(v16h a, v16h b, v8f c) {
  // D = A(16x32 f16) * B(32x16 f16) + C(16x16 f32)
  return __builtin_amdgcn_wmma_f32_16x16x32_f16(
      /*neg_a=*/false, a, /*neg_b=*/false, b,
      /*c_mod=*/(short)0, c, /*reuse_a=*/false, /*reuse_b=*/false);
}

// Async global->LDS copy, 16 bytes per lane.  Tracked by ASYNCcnt.
DEVINL void async_ld_b128(unsigned lds_byte_addr, const void* gsrc) {
  asm volatile("global_load_async_to_lds_b128 %0, %1, off"
               :
               : "v"(lds_byte_addr), "v"(gsrc)
               : "memory");
}

DEVINL void wait_async0() {
  asm volatile("s_wait_asynccnt 0x0" ::: "memory");
}

DEVINL unsigned lds_addr_of(const void* p) {
  return (unsigned)(uintptr_t)p;  // LDS address is the low 32 bits of the generic pointer (ISA 10.2)
}

// A-fragment (16x32, f16): lane L holds row (L&15); elems 0..7 -> K = h*8+e,
// elems 8..15 -> K = 16 + h*8 + e  (h = L>>4).  Source row-major, ld in elems.
DEVINL v16h load_a_f16(const half_t* base, int row0, int ld, int k0) {
  const unsigned lane = threadIdx.x & 31u;
  const int r = lane & 15, h = lane >> 4;
  const half_t* p = base + (size_t)(row0 + r) * (size_t)ld + k0 + h * 8;
  v8h lo = *reinterpret_cast<const v8h*>(p);
  v8h hi = *reinterpret_cast<const v8h*>(p + 16);
  v16h f;
#pragma unroll
  for (int i = 0; i < 8; ++i) { f[i] = lo[i]; f[8 + i] = hi[i]; }
  return f;
}

DEVINL v16h load_a_f32(const float* base, int row0, int ld, int k0) {
  const unsigned lane = threadIdx.x & 31u;
  const int r = lane & 15, h = lane >> 4;
  const float* p = base + (size_t)(row0 + r) * (size_t)ld + k0 + h * 8;
  v4f a0 = *reinterpret_cast<const v4f*>(p);
  v4f a1 = *reinterpret_cast<const v4f*>(p + 4);
  v4f b0 = *reinterpret_cast<const v4f*>(p + 16);
  v4f b1 = *reinterpret_cast<const v4f*>(p + 20);
  v16h f;
#pragma unroll
  for (int i = 0; i < 4; ++i) {
    f[i]      = (half_t)a0[i];
    f[4 + i]  = (half_t)a1[i];
    f[8 + i]  = (half_t)b0[i];
    f[12 + i] = (half_t)b1[i];
  }
  return f;
}

// B-fragment (32x16, f16): lane L holds column (L&15); elems 0..7 -> K = h*8+e,
// elems 8..15 -> K = 16 + h*8 + e.
// Source row-major over the matrix *column* index, K contiguous.  ld in elems.
DEVINL v16h load_b_f16(const half_t* base, int col0, int ld, int k0) {
  const unsigned lane = threadIdx.x & 31u;
  const int n = lane & 15, h = lane >> 4;
  const half_t* p = base + (size_t)(col0 + n) * (size_t)ld + k0 + h * 8;
  v8h lo = *reinterpret_cast<const v8h*>(p);
  v8h hi = *reinterpret_cast<const v8h*>(p + 16);
  v16h f;
#pragma unroll
  for (int i = 0; i < 8; ++i) { f[i] = lo[i]; f[8 + i] = hi[i]; }
  return f;
}

// Cooperative (256-thread) stage of a 64x32 f32 weight tile into LDS as f16.
// wbuf layout: [64 rows][32 cols] f16, row = output channel, col = K.
DEVINL void stage_w_tile(const float* __restrict__ W, int n0, int k0,
                         half_t* wbuf) {
  const int t = threadIdx.x;       // 0..255
  const int row = t >> 2;          // 0..63
  const int c0 = (t & 3) * 8;      // 0,8,16,24
  const float* src = W + (size_t)(n0 + row) * kC + k0 + c0;
  v4f f0 = *reinterpret_cast<const v4f*>(src);
  v4f f1 = *reinterpret_cast<const v4f*>(src + 4);
  v8h hv;
#pragma unroll
  for (int i = 0; i < 4; ++i) {
    hv[i]     = (half_t)f0[i];
    hv[4 + i] = (half_t)f1[i];
  }
  *reinterpret_cast<v8h*>(wbuf + row * 32 + c0) = hv;
}

// ---------------- Kernel 1: fused QKV projection ----------------
// y = x @ W.T + b.  grid (M/128, C/64, 3), block 256 (8 waves).
// Weight tiles double-buffered in LDS: tile k0+32 is loaded/converted/stored
// while tile k0 feeds the WMMAs.
// mode 0: q (pre-scaled) -> (b,h,t,d); mode 1: k -> (b,h,t,d);
// mode 2: v -> (b,h,d,t) transposed for the PV WMMA B-fragment.
__global__ __launch_bounds__(256) void qkv_proj_kernel(
    const float* __restrict__ x,
    const float* __restrict__ Wq, const float* __restrict__ bq,
    const float* __restrict__ Wk, const float* __restrict__ bk,
    const float* __restrict__ Wv, const float* __restrict__ bv,
    const float* __restrict__ attn_scale,
    half_t* __restrict__ qws, half_t* __restrict__ kws,
    half_t* __restrict__ vws) {
  __shared__ __align__(16) half_t wbuf[2][64 * 32];  // 2 x 4 KB f16 tiles

  const int mode = blockIdx.z;
  const float* W    = (mode == 0) ? Wq : (mode == 1) ? Wk : Wv;
  const float* bias = (mode == 0) ? bq : (mode == 1) ? bk : bv;

  const int wid = threadIdx.x >> 5;
  const int m0 = blockIdx.x * 128 + wid * 16;
  const int n0 = blockIdx.y * 64;

  stage_w_tile(W, n0, 0, wbuf[0]);  // prologue: tile 0

  v8f acc[4] = {{}, {}, {}, {}};
  for (int k0 = 0, i = 0; k0 < kC; k0 += 32, ++i) {
    __syncthreads();  // tile i stores complete + prev buffer reads retired
    if (k0 + 32 < kC) stage_w_tile(W, n0, k0 + 32, wbuf[(i + 1) & 1]);
    const half_t* wb = wbuf[i & 1];
    v16h a = load_a_f32(x, m0, kC, k0);
#pragma unroll
    for (int j = 0; j < 4; ++j) {
      v16h bf = load_b_f16(wb, j * 16, 32, 0);
      acc[j] = wmma_f16(a, bf, acc[j]);
    }
  }

  const unsigned lane = threadIdx.x & 31u;
  const int c = lane & 15, h = lane >> 4;
  const float sc = attn_scale[0];
#pragma unroll
  for (int j = 0; j < 4; ++j) {
#pragma unroll
    for (int r = 0; r < 8; ++r) {
      const int m = m0 + r + 8 * h;        // global row in [0, B*T)
      const int n = n0 + j * 16 + c;       // channel in [0, C)
      const int b = m / kT, t = m - b * kT;
      const int hd = n >> 6, dd = n & 63;
      float val = acc[j][r] + bias[n];
      if (mode == 0) {
        qws[(size_t)(((b * kH + hd) * kT + t) * kD + dd)] = (half_t)(val * sc);
      } else if (mode == 1) {
        kws[(size_t)(((b * kH + hd) * kT + t) * kD + dd)] = (half_t)val;
      } else {
        vws[(size_t)(((b * kH + hd) * kD + dd) * kT + t)] = (half_t)val;
      }
    }
  }
}

// ---------------- Kernel 2: flash attention with chunk mask ----------------
// Mask (derived from reference): a 64-query block attends fully to later
// 64-blocks, causally (k<=q) within its own block, never to earlier blocks.
// grid (B*H, T/64), block 128 (4 waves = one 64-query block).  All waves share
// the same key range; K/V slabs are staged block-wide with
// global_load_async_to_lds_b128 (ASYNCcnt), double-buffered so slab i+1's DMA
// overlaps slab i's WMMAs + softmax.
__global__ __launch_bounds__(128) void attn_kernel(
    const half_t* __restrict__ qws, const half_t* __restrict__ kws,
    const half_t* __restrict__ vws, half_t* __restrict__ ows) {
  __shared__ __align__(16) half_t kbuf[2][32 * 64];   // keys: 32 rows x D
  __shared__ __align__(16) half_t vbuf[2][64 * 32];   // vT:   D rows x 32 keys
  __shared__ __align__(16) half_t pbuf[4][16 * 32];   // per-wave P staging

  const int tid = threadIdx.x;
  const int wid = tid >> 5;
  const unsigned lane = tid & 31u;
  const int c = lane & 15, h = lane >> 4;

  const int bh = blockIdx.x;               // b*H + head
  const int b = bh / kH, head = bh - b * kH;
  const int blk0 = blockIdx.y * kChunk;    // 64-query block base
  const int qt0 = blk0 + wid * 16;         // this wave's query strip

  const half_t* qbase = qws + (size_t)bh * kT * kD;
  const half_t* kbase = kws + (size_t)bh * kT * kD;
  const half_t* vbase = vws + (size_t)bh * kD * kT;  // (d, t) layout

  // Q fragments kept live across the whole key loop (D=64 -> two K=32 halves)
  const v16h aq0 = load_a_f16(qbase, qt0, kD, 0);
  const v16h aq1 = load_a_f16(qbase, qt0, kD, 32);

  float mrun[8], lrun[8];
  v8f oacc[4] = {{}, {}, {}, {}};
#pragma unroll
  for (int r = 0; r < 8; ++r) { mrun[r] = kNegInf; lrun[r] = 0.0f; }

  half_t* myp = pbuf[wid];

  // Per-thread async-copy geometry (constant per iteration):
  // K slab: contiguous 4KB; thread copies 32B at byte offset tid*32.
  // V slab: 64 rows x 64B; thread copies 32B of row (tid>>1), half (tid&1).
  const unsigned kslab_off = (unsigned)tid * 32u;
  const unsigned vslab_off =
      (unsigned)(((tid >> 1) * 32 + (tid & 1) * 16) * 2);
  const half_t* vsrc_row = vbase + (size_t)(tid >> 1) * kT + (tid & 1) * 16;

  auto issue_slab = [&](int n0, int sel) {
    const half_t* ksrc = kbase + (size_t)n0 * kD + tid * 16;
    const unsigned kl = lds_addr_of(kbuf[sel]) + kslab_off;
    async_ld_b128(kl,       ksrc);
    async_ld_b128(kl + 16u, ksrc + 8);
    const half_t* vsrc = vsrc_row + n0;
    const unsigned vl = lds_addr_of(vbuf[sel]) + vslab_off;
    async_ld_b128(vl,       vsrc);
    async_ld_b128(vl + 16u, vsrc + 8);
  };

  issue_slab(blk0, 0);  // prologue: slab 0 in flight

  const int n_iter = (kT - blk0) >> 5;
  for (int i = 0; i < n_iter; ++i) {
    const int n0 = blk0 + i * 32;

    wait_async0();    // slab i landed in LDS (only slab i is outstanding here)
    __syncthreads();  // visible block-wide; old buffer's readers retired
    if (i + 1 < n_iter) issue_slab(n0 + 32, (i + 1) & 1);  // overlap with math

    const half_t* kb = kbuf[i & 1];
    const half_t* vb = vbuf[i & 1];

    // ---- S = (q*scale) @ K^T for a 16x32 key slab ----
    v8f s0 = {}, s1 = {};
    {
      v16h k00 = load_b_f16(kb, 0, kD, 0);
      v16h k01 = load_b_f16(kb, 0, kD, 32);
      s0 = wmma_f16(aq0, k00, s0);
      s0 = wmma_f16(aq1, k01, s0);
      v16h k10 = load_b_f16(kb, 16, kD, 0);
      v16h k11 = load_b_f16(kb, 16, kD, 32);
      s1 = wmma_f16(aq0, k10, s1);
      s1 = wmma_f16(aq1, k11, s1);
    }

    // ---- diagonal-block causal mask ----
    if (n0 - blk0 < kChunk) {
      const int kg0 = n0 + c, kg1 = n0 + 16 + c;
#pragma unroll
      for (int r = 0; r < 8; ++r) {
        const int qg = qt0 + r + 8 * h;
        if (kg0 > qg) s0[r] = kNegInf;
        if (kg1 > qg) s1[r] = kNegInf;
      }
    }

    // ---- online softmax (row stats per-lane, replicated over 16 lanes) ----
    float mt[8];
#pragma unroll
    for (int r = 0; r < 8; ++r) mt[r] = fmaxf(s0[r], s1[r]);
#pragma unroll
    for (int off = 1; off < 16; off <<= 1)
#pragma unroll
      for (int r = 0; r < 8; ++r) mt[r] = fmaxf(mt[r], __shfl_xor(mt[r], off, 32));

    float p0[8], p1[8], rs[8], al[8];
#pragma unroll
    for (int r = 0; r < 8; ++r) {
      const float mn = fmaxf(mrun[r], mt[r]);
      al[r] = __expf(mrun[r] - mn);
      mrun[r] = mn;
      p0[r] = __expf(s0[r] - mn);
      p1[r] = __expf(s1[r] - mn);
      rs[r] = p0[r] + p1[r];
    }
#pragma unroll
    for (int off = 1; off < 16; off <<= 1)
#pragma unroll
      for (int r = 0; r < 8; ++r) rs[r] += __shfl_xor(rs[r], off, 32);
#pragma unroll
    for (int r = 0; r < 8; ++r) {
      lrun[r] = lrun[r] * al[r] + rs[r];
#pragma unroll
      for (int j = 0; j < 4; ++j) oacc[j][r] *= al[r];
    }

    // ---- re-layout P (C-layout -> A-fragment) through wave-private LDS ----
#pragma unroll
    for (int r = 0; r < 8; ++r) {
      myp[(r + 8 * h) * 32 + c]      = (half_t)p0[r];
      myp[(r + 8 * h) * 32 + 16 + c] = (half_t)p1[r];
    }
    asm volatile("s_wait_dscnt 0" ::: "memory");  // wave-local LDS RAW fence
    const v16h pa = load_a_f16(myp, 0, 32, 0);

    // ---- O += P @ V  (V^T slab in LDS: B-fragment is contiguous) ----
#pragma unroll
    for (int j = 0; j < 4; ++j) {
      v16h bv = load_b_f16(vb, j * 16, 32, 0);
      oacc[j] = wmma_f16(pa, bv, oacc[j]);
    }
  }

  // ---- finalize: divide by row sums, store (b, t, h*D+d) f16 ----
#pragma unroll
  for (int j = 0; j < 4; ++j) {
#pragma unroll
    for (int r = 0; r < 8; ++r) {
      const int t = qt0 + r + 8 * h;
      const int n = head * kD + j * 16 + c;
      ows[(size_t)(b * kT + t) * kC + n] = (half_t)(oacc[j][r] / lrun[r]);
    }
  }
}

// ---------------- Kernel 3: output projection (f32 result) ----------------
__global__ __launch_bounds__(256) void out_proj_kernel(
    const half_t* __restrict__ ows, const float* __restrict__ Wo,
    const float* __restrict__ bo, float* __restrict__ out) {
  __shared__ __align__(16) half_t wbuf[2][64 * 32];

  const int wid = threadIdx.x >> 5;
  const int m0 = blockIdx.x * 128 + wid * 16;
  const int n0 = blockIdx.y * 64;

  stage_w_tile(Wo, n0, 0, wbuf[0]);

  v8f acc[4] = {{}, {}, {}, {}};
  for (int k0 = 0, i = 0; k0 < kC; k0 += 32, ++i) {
    __syncthreads();
    if (k0 + 32 < kC) stage_w_tile(Wo, n0, k0 + 32, wbuf[(i + 1) & 1]);
    const half_t* wb = wbuf[i & 1];
    v16h a = load_a_f16(ows, m0, kC, k0);
#pragma unroll
    for (int j = 0; j < 4; ++j) {
      v16h bf = load_b_f16(wb, j * 16, 32, 0);
      acc[j] = wmma_f16(a, bf, acc[j]);
    }
  }

  const unsigned lane = threadIdx.x & 31u;
  const int c = lane & 15, h = lane >> 4;
#pragma unroll
  for (int j = 0; j < 4; ++j) {
#pragma unroll
    for (int r = 0; r < 8; ++r) {
      const int m = m0 + r + 8 * h;
      const int n = n0 + j * 16 + c;
      out[(size_t)m * kC + n] = acc[j][r] + bo[n];
    }
  }
}

// ---------------- host launcher ----------------
extern "C" void kernel_launch(void* const* d_in, const int* in_sizes, int n_in,
                              void* d_out, int out_size, void* d_ws,
                              size_t ws_size, hipStream_t stream) {
  (void)in_sizes; (void)n_in; (void)out_size; (void)ws_size;
  const float* x  = (const float*)d_in[0];
  const float* Wq = (const float*)d_in[1];
  const float* bq = (const float*)d_in[2];
  const float* Wk = (const float*)d_in[3];
  const float* bk = (const float*)d_in[4];
  const float* Wv = (const float*)d_in[5];
  const float* bv = (const float*)d_in[6];
  const float* Wo = (const float*)d_in[7];
  const float* bo = (const float*)d_in[8];
  const float* sc = (const float*)d_in[9];
  float* out = (float*)d_out;

  char* ws = (char*)d_ws;
  const size_t seg = (size_t)kB * kT * kC * sizeof(half_t);  // 8 MiB
  half_t* qws = (half_t*)(ws + 0 * seg);
  half_t* kws = (half_t*)(ws + 1 * seg);
  half_t* vws = (half_t*)(ws + 2 * seg);
  half_t* ows = (half_t*)(ws + 3 * seg);

  const int M = kB * kT;  // 4096
  dim3 g1(M / 128, kC / 64, 3);
  qkv_proj_kernel<<<g1, 256, 0, stream>>>(x, Wq, bq, Wk, bk, Wv, bv, sc,
                                          qws, kws, vws);

  dim3 g2(kB * kH, kT / 64);
  attn_kernel<<<g2, 128, 0, stream>>>(qws, kws, vws, ows);

  dim3 g3(M / 128, kC / 64);
  out_proj_kernel<<<g3, 256, 0, stream>>>(ows, Wo, bo, out);
}
